// MoEVAE_82420422410528
// MI455X (gfx1250) — matrix operand order repaired and checked by reading, not compiled, not run on hardware
//
#include <hip/hip_runtime.h>

// ---------------- types ----------------
typedef __bf16 bf16;
typedef __attribute__((ext_vector_type(16))) __bf16 v16bf;
typedef __attribute__((ext_vector_type(8)))  float  v8f;
typedef int v4i __attribute__((vector_size(16)));

// ---------------- problem constants ----------------
#define BATCH 16384
#define D_IN  1024
#define H1_N  2048
#define H2_N  1024
#define E_N   16
#define L_N   256
#define K_TOP 2
#define D1_N  1024
#define D2_N  2048

// d_out layout (floats): recon | mu | logvar | probs
#define OFF_RECON 0
#define OFF_MU    ((size_t)BATCH * D_IN)
#define OFF_LV    (OFF_MU + (size_t)BATCH * E_N * L_N)
#define OFF_PROBS (OFF_LV + (size_t)BATCH * E_N * L_N)

// ---------------- CDNA5 async global->LDS support (guarded) ----------------
#if __has_builtin(__builtin_amdgcn_global_load_async_to_lds_b128)
#define HAVE_ASYNC_LDS 1
__device__ __forceinline__ void async_cp16(void* lds, const void* g) {
  // prototype (from hipcc diagnostic): (v4i AS1*, v4i AS3*, Imm offset, Imm cpol)
  __builtin_amdgcn_global_load_async_to_lds_b128(
      (__attribute__((address_space(1))) v4i*)(g),
      (__attribute__((address_space(3))) v4i*)(lds), 0, 0);
}
#else
#define HAVE_ASYNC_LDS 0
#endif

__device__ __forceinline__ void wait_async0() {
#if __has_builtin(__builtin_amdgcn_s_wait_asynccnt)
  __builtin_amdgcn_s_wait_asynccnt(0);
#else
  asm volatile("s_wait_asynccnt 0x0" ::: "memory");
#endif
}

// ---------------- fp32 -> bf16 elementwise ----------------
__global__ __launch_bounds__(256) void cvt_bf16_kernel(const float* __restrict__ in,
                                                       bf16* __restrict__ out, size_t n) {
  size_t i = (size_t)blockIdx.x * 256 + threadIdx.x;
  if (i < n) out[i] = (bf16)in[i];
}

// fp32 [K,N] -> bf16 [N,K] (transpose + convert); done once per launch, tiny cost
__global__ __launch_bounds__(256) void cvt_bf16_T_kernel(const float* __restrict__ in,
                                                         bf16* __restrict__ out, int K, int N) {
  size_t i = (size_t)blockIdx.x * 256 + threadIdx.x;
  size_t total = (size_t)K * N;
  if (i >= total) return;
  int k = (int)(i / N);
  int n = (int)(i % N);
  out[(size_t)n * K + k] = (bf16)in[i];
}

// ---------------- WMMA bf16 GEMM: C[M,N] = A[M,K] @ Bt[N,K]^T + bias ----------------
// 128x128 block tile, BK=32, 8 waves (4x2), 32x64 per wave, double-buffered LDS.
#define BM 128
#define BN 128
#define BKT 32
#define LDSP 40  // padded stride in bf16 elems: 80B rows, keeps 16B alignment

struct FragU { union { v16bf v; uint4 q[2]; }; };

__global__ __launch_bounds__(256) void gemm_bias_kernel(const bf16* __restrict__ A,
                                                        const bf16* __restrict__ Bt,
                                                        const float* __restrict__ bias,
                                                        float* __restrict__ C,
                                                        int M, int N, int K) {
  __shared__ bf16 Asl[2][BM * LDSP];
  __shared__ bf16 Bsl[2][BN * LDSP];

  const int tid  = threadIdx.x;
  const int lane = tid & 31;
  const int wid  = tid >> 5;
  const int wm   = wid & 3;   // 0..3 -> 32 rows each
  const int wn   = wid >> 2;  // 0..1 -> 64 cols each
  const int h    = lane >> 4; // half-wave select
  const int l16  = lane & 15;
  const int m0   = blockIdx.y * BM;
  const int n0   = blockIdx.x * BN;

  // per-thread copy coordinates: 2 chunks of 8 bf16 (16B) for each of A and B
  int rowc[2], c8c[2];
#pragma unroll
  for (int it = 0; it < 2; it++) {
    int idx = tid + it * 256;    // 0..511
    rowc[it] = idx >> 2;         // 0..127
    c8c[it]  = (idx & 3) << 3;   // 0,8,16,24
  }

  v8f acc[2][4];
#pragma unroll
  for (int i = 0; i < 2; i++)
#pragma unroll
    for (int j = 0; j < 4; j++) {
      v8f z = {0.f, 0.f, 0.f, 0.f, 0.f, 0.f, 0.f, 0.f};
      acc[i][j] = z;
    }

  const int KT = K / BKT;

  // ---- prologue: stage tile 0 into buffer 0 ----
#if HAVE_ASYNC_LDS
#pragma unroll
  for (int it = 0; it < 2; it++) {
    async_cp16(&Asl[0][rowc[it] * LDSP + c8c[it]], &A[(size_t)(m0 + rowc[it]) * K + c8c[it]]);
    async_cp16(&Bsl[0][rowc[it] * LDSP + c8c[it]], &Bt[(size_t)(n0 + rowc[it]) * K + c8c[it]]);
  }
  wait_async0();
#else
#pragma unroll
  for (int it = 0; it < 2; it++) {
    *(uint4*)&Asl[0][rowc[it] * LDSP + c8c[it]] = *(const uint4*)&A[(size_t)(m0 + rowc[it]) * K + c8c[it]];
    *(uint4*)&Bsl[0][rowc[it] * LDSP + c8c[it]] = *(const uint4*)&Bt[(size_t)(n0 + rowc[it]) * K + c8c[it]];
  }
#endif
  __syncthreads();

  for (int kt = 0; kt < KT; kt++) {
    const int cur = kt & 1;
    const int nxt = cur ^ 1;
    const int k1  = (kt + 1) * BKT;
    const bool has_next = (kt + 1) < KT;

#if HAVE_ASYNC_LDS
    if (has_next) {
#pragma unroll
      for (int it = 0; it < 2; it++) {
        async_cp16(&Asl[nxt][rowc[it] * LDSP + c8c[it]], &A[(size_t)(m0 + rowc[it]) * K + k1 + c8c[it]]);
        async_cp16(&Bsl[nxt][rowc[it] * LDSP + c8c[it]], &Bt[(size_t)(n0 + rowc[it]) * K + k1 + c8c[it]]);
      }
    }
#else
    uint4 ra[2], rb[2];
    if (has_next) {
#pragma unroll
      for (int it = 0; it < 2; it++) {
        ra[it] = *(const uint4*)&A[(size_t)(m0 + rowc[it]) * K + k1 + c8c[it]];
        rb[it] = *(const uint4*)&Bt[(size_t)(n0 + rowc[it]) * K + k1 + c8c[it]];
      }
    }
#endif

    // ---- compute on buffer `cur` ----
    FragU afr[2], bfr[4];
#pragma unroll
    for (int i = 0; i < 2; i++) {
      const bf16* p = &Asl[cur][(wm * 32 + i * 16 + l16) * LDSP];
      // A 16x32 bf16 layout: lane half h holds K = {h*8..h*8+7} and {16+h*8..+7}
      afr[i].q[0] = *(const uint4*)(p + h * 8);
      afr[i].q[1] = *(const uint4*)(p + 16 + h * 8);
    }
#pragma unroll
    for (int j = 0; j < 4; j++) {
      const bf16* p = &Bsl[cur][(wn * 64 + j * 16 + l16) * LDSP];
      // B 32x16 bf16 layout: lane = column, half h holds K = {h*8..h*8+7} and {16+h*8..+7}
      bfr[j].q[0] = *(const uint4*)(p + h * 8);
      bfr[j].q[1] = *(const uint4*)(p + 16 + h * 8);
    }
#pragma unroll
    for (int i = 0; i < 2; i++)
#pragma unroll
      for (int j = 0; j < 4; j++)
        acc[i][j] = __builtin_amdgcn_wmma_f32_16x16x32_bf16(
            false, afr[i].v, false, bfr[j].v, (short)0, acc[i][j], false, false);

#if HAVE_ASYNC_LDS
    wait_async0();
#else
    if (has_next) {
#pragma unroll
      for (int it = 0; it < 2; it++) {
        *(uint4*)&Asl[nxt][rowc[it] * LDSP + c8c[it]] = ra[it];
        *(uint4*)&Bsl[nxt][rowc[it] * LDSP + c8c[it]] = rb[it];
      }
    }
#endif
    __syncthreads();
  }

  // ---- epilogue: bias add, fp32 store. D layout: row = r + 8*h, col = l16 ----
#pragma unroll
  for (int j = 0; j < 4; j++) {
    int gn = n0 + wn * 64 + j * 16 + l16;
    float bv = bias[gn];
#pragma unroll
    for (int i = 0; i < 2; i++) {
#pragma unroll
      for (int r = 0; r < 8; r++) {
        int gm = m0 + wm * 32 + i * 16 + h * 8 + r;
        C[(size_t)gm * N + gn] = acc[i][j][r] + bv;
      }
    }
  }
}

// ---------------- LayerNorm + ReLU -> bf16, one block per row ----------------
__global__ __launch_bounds__(256) void ln_relu_kernel(const float* __restrict__ X,
                                                      const float* __restrict__ g,
                                                      const float* __restrict__ be,
                                                      bf16* __restrict__ Y, int N) {
  __shared__ float red[256];
  int row = blockIdx.x;
  const float* x = X + (size_t)row * N;
  float s = 0.f, ss = 0.f;
  for (int i = threadIdx.x; i < N; i += 256) {
    float v = x[i];
    s += v;
    ss += v * v;
  }
  red[threadIdx.x] = s;
  __syncthreads();
  for (int st = 128; st > 0; st >>= 1) {
    if (threadIdx.x < st) red[threadIdx.x] += red[threadIdx.x + st];
    __syncthreads();
  }
  float mean = red[0] / (float)N;
  __syncthreads();
  red[threadIdx.x] = ss;
  __syncthreads();
  for (int st = 128; st > 0; st >>= 1) {
    if (threadIdx.x < st) red[threadIdx.x] += red[threadIdx.x + st];
    __syncthreads();
  }
  float var  = red[0] / (float)N - mean * mean;
  float rstd = rsqrtf(var + 1e-5f);
  for (int i = threadIdx.x; i < N; i += 256) {
    float v = (x[i] - mean) * rstd * g[i] + be[i];
    v = v > 0.f ? v : 0.f;
    Y[(size_t)row * N + i] = (bf16)v;
  }
}

// ---------------- router small GEMM: logits[b,e] = h2[b,:] . Wrt[e,:] + br[e] ------
union U8bf { uint4 q; bf16 hh[8]; };

__global__ __launch_bounds__(256) void router_gemm_kernel(const bf16* __restrict__ h2,
                                                          const bf16* __restrict__ Wrt,
                                                          const float* __restrict__ br,
                                                          float* __restrict__ out) {
  int idx = blockIdx.x * 256 + threadIdx.x;  // BATCH*16
  int b = idx >> 4, e = idx & 15;
  const bf16* hp = h2 + (size_t)b * H2_N;
  const bf16* wp = Wrt + (size_t)e * H2_N;
  float acc = 0.f;
  for (int k = 0; k < H2_N; k += 8) {
    U8bf a, w;
    a.q = *(const uint4*)(hp + k);
    w.q = *(const uint4*)(wp + k);
#pragma unroll
    for (int j = 0; j < 8; j++) acc += (float)a.hh[j] * (float)w.hh[j];
  }
  out[idx] = acc + br[e];
}

// ---------------- router LN + softmax + top-2, one thread per row ----------------
__global__ __launch_bounds__(256) void router_epi_kernel(const float* __restrict__ logits,
                                                         const float* __restrict__ g,
                                                         const float* __restrict__ be,
                                                         float* __restrict__ probs,
                                                         float* __restrict__ t2v,
                                                         int* __restrict__ t2i, int B) {
  int b = blockIdx.x * 256 + threadIdx.x;
  if (b >= B) return;
  float v[E_N];
  float s = 0.f, ss = 0.f;
  for (int e = 0; e < E_N; e++) {
    v[e] = logits[(size_t)b * E_N + e];
    s += v[e];
    ss += v[e] * v[e];
  }
  float m = s / (float)E_N;
  float var = ss / (float)E_N - m * m;
  float rstd = rsqrtf(var + 1e-5f);
  float mx = -3.4e38f;
  for (int e = 0; e < E_N; e++) {
    v[e] = (v[e] - m) * rstd * g[e] + be[e];
    mx = fmaxf(mx, v[e]);
  }
  float den = 0.f;
  for (int e = 0; e < E_N; e++) {
    v[e] = expf(v[e] - mx);
    den += v[e];
  }
  float inv = 1.f / den;
  int i0 = 0;
  float p0 = -1.f;
  for (int e = 0; e < E_N; e++) {
    v[e] *= inv;
    probs[(size_t)b * E_N + e] = v[e];
    if (v[e] > p0) { p0 = v[e]; i0 = e; }
  }
  int i1 = 0;
  float p1 = -1.f;
  for (int e = 0; e < E_N; e++)
    if (e != i0 && v[e] > p1) { p1 = v[e]; i1 = e; }
  t2v[b * 2 + 0] = p0;
  t2v[b * 2 + 1] = p1;
  t2i[b * 2 + 0] = i0;
  t2i[b * 2 + 1] = i1;
}

// ---------------- reparameterize + gate combine -> z (bf16) ----------------
__global__ __launch_bounds__(256) void z_kernel(const float* __restrict__ mu,
                                                const float* __restrict__ lv,
                                                const float* __restrict__ noise,
                                                const float* __restrict__ t2v,
                                                const int* __restrict__ t2i,
                                                bf16* __restrict__ zb) {
  int idx = blockIdx.x * 256 + threadIdx.x;  // BATCH*L_N
  int b = idx >> 8, l = idx & 255;
  float z = 0.f;
#pragma unroll
  for (int k = 0; k < K_TOP; k++) {
    int e = t2i[b * 2 + k];
    float w = t2v[b * 2 + k];
    size_t o = (size_t)b * (E_N * L_N) + (size_t)e * L_N + l;
    float mk = mu[o];
    float lk = lv[o];
    float nz = noise[((size_t)b * K_TOP + k) * L_N + l];
    z += (mk + nz * expf(0.5f * lk)) * w;
  }
  zb[idx] = (bf16)z;
}

// ---------------- host-side launch ----------------
static inline size_t alignup(size_t x) { return (x + 255) & ~(size_t)255; }

extern "C" void kernel_launch(void* const* d_in, const int* in_sizes, int n_in,
                              void* d_out, int out_size, void* d_ws, size_t ws_size,
                              hipStream_t stream) {
  (void)in_sizes; (void)n_in; (void)out_size; (void)ws_size;
  const float* x    = (const float*)d_in[0];
  const float* nois = (const float*)d_in[1];
  const float* W_e1 = (const float*)d_in[2];  const float* b_e1 = (const float*)d_in[3];
  const float* g_e1 = (const float*)d_in[4];  const float* be_e1= (const float*)d_in[5];
  const float* W_e2 = (const float*)d_in[6];  const float* b_e2 = (const float*)d_in[7];
  const float* g_e2 = (const float*)d_in[8];  const float* be_e2= (const float*)d_in[9];
  const float* W_r  = (const float*)d_in[10]; const float* b_r  = (const float*)d_in[11];
  const float* g_r  = (const float*)d_in[12]; const float* be_r = (const float*)d_in[13];
  const float* W_mu = (const float*)d_in[14]; const float* b_mu = (const float*)d_in[15];
  const float* W_lv = (const float*)d_in[16]; const float* b_lv = (const float*)d_in[17];
  const float* W_d1 = (const float*)d_in[18]; const float* b_d1 = (const float*)d_in[19];
  const float* g_d1 = (const float*)d_in[20]; const float* be_d1= (const float*)d_in[21];
  const float* W_d2 = (const float*)d_in[22]; const float* b_d2 = (const float*)d_in[23];
  const float* g_d2 = (const float*)d_in[24]; const float* be_d2= (const float*)d_in[25];
  const float* W_do = (const float*)d_in[26]; const float* b_do = (const float*)d_in[27];

  float* out = (float*)d_out;
  char*  ws  = (char*)d_ws;

  // workspace carve-up (bytes)
  size_t off = 0;
  size_t OFF_T    = off; off = alignup(off + (size_t)BATCH * 2048 * 4);      // fp32 pre-LN scratch
  size_t OFF_BUFA = off; off = alignup(off + (size_t)BATCH * 2048 * 2);      // h1b / d2b (bf16)
  size_t OFF_BUFB = off; off = alignup(off + (size_t)BATCH * 1024 * 2);      // xb  / d1b (bf16)
  size_t OFF_H2B  = off; off = alignup(off + (size_t)BATCH * 1024 * 2);      // h2b (bf16)
  size_t OFF_ZB   = off; off = alignup(off + (size_t)BATCH * L_N  * 2);      // z (bf16)
  size_t OFF_RLOG = off; off = alignup(off + (size_t)BATCH * E_N  * 4);      // router logits
  size_t OFF_T2V  = off; off = alignup(off + (size_t)BATCH * 2 * 4);
  size_t OFF_T2I  = off; off = alignup(off + (size_t)BATCH * 2 * 4);
  size_t OFF_WE1  = off; off = alignup(off + (size_t)D_IN * H1_N * 2);
  size_t OFF_WE2  = off; off = alignup(off + (size_t)H1_N * H2_N * 2);
  size_t OFF_WR   = off; off = alignup(off + (size_t)H2_N * E_N * 2);
  size_t OFF_WMU  = off; off = alignup(off + (size_t)H2_N * E_N * L_N * 2);
  size_t OFF_WLV  = off; off = alignup(off + (size_t)H2_N * E_N * L_N * 2);
  size_t OFF_WD1  = off; off = alignup(off + (size_t)L_N * D1_N * 2);
  size_t OFF_WD2  = off; off = alignup(off + (size_t)D1_N * D2_N * 2);
  size_t OFF_WDO  = off; off = alignup(off + (size_t)D2_N * D_IN * 2);

  float* t     = (float*)(ws + OFF_T);
  bf16*  bufA  = (bf16*)(ws + OFF_BUFA);   // h1b then d2b
  bf16*  bufB  = (bf16*)(ws + OFF_BUFB);   // xb then d1b
  bf16*  h2b   = (bf16*)(ws + OFF_H2B);
  bf16*  zb    = (bf16*)(ws + OFF_ZB);
  float* rlog  = (float*)(ws + OFF_RLOG);
  float* t2v   = (float*)(ws + OFF_T2V);
  int*   t2i   = (int*)(ws + OFF_T2I);
  bf16*  We1t  = (bf16*)(ws + OFF_WE1);
  bf16*  We2t  = (bf16*)(ws + OFF_WE2);
  bf16*  Wrt   = (bf16*)(ws + OFF_WR);
  bf16*  Wmut  = (bf16*)(ws + OFF_WMU);
  bf16*  Wlvt  = (bf16*)(ws + OFF_WLV);
  bf16*  Wd1t  = (bf16*)(ws + OFF_WD1);
  bf16*  Wd2t  = (bf16*)(ws + OFF_WD2);
  bf16*  Wdot  = (bf16*)(ws + OFF_WDO);

  dim3 blk(256);
  auto g1 = [](size_t n) { return dim3((unsigned)((n + 255) / 256)); };

  // --- precision prep: x -> bf16, weights -> bf16 transposed [N,K] ---
  hipLaunchKernelGGL(cvt_bf16_kernel, g1((size_t)BATCH * D_IN), blk, 0, stream, x, bufB, (size_t)BATCH * D_IN);
  hipLaunchKernelGGL(cvt_bf16_T_kernel, g1((size_t)D_IN * H1_N), blk, 0, stream, W_e1, We1t, D_IN, H1_N);
  hipLaunchKernelGGL(cvt_bf16_T_kernel, g1((size_t)H1_N * H2_N), blk, 0, stream, W_e2, We2t, H1_N, H2_N);
  hipLaunchKernelGGL(cvt_bf16_T_kernel, g1((size_t)H2_N * E_N),  blk, 0, stream, W_r,  Wrt,  H2_N, E_N);
  hipLaunchKernelGGL(cvt_bf16_T_kernel, g1((size_t)H2_N * E_N * L_N), blk, 0, stream, W_mu, Wmut, H2_N, E_N * L_N);
  hipLaunchKernelGGL(cvt_bf16_T_kernel, g1((size_t)H2_N * E_N * L_N), blk, 0, stream, W_lv, Wlvt, H2_N, E_N * L_N);
  hipLaunchKernelGGL(cvt_bf16_T_kernel, g1((size_t)L_N * D1_N),  blk, 0, stream, W_d1, Wd1t, L_N, D1_N);
  hipLaunchKernelGGL(cvt_bf16_T_kernel, g1((size_t)D1_N * D2_N), blk, 0, stream, W_d2, Wd2t, D1_N, D2_N);
  hipLaunchKernelGGL(cvt_bf16_T_kernel, g1((size_t)D2_N * D_IN), blk, 0, stream, W_do, Wdot, D2_N, D_IN);

  // --- encoder ---
  hipLaunchKernelGGL(gemm_bias_kernel, dim3(H1_N / BN, BATCH / BM), blk, 0, stream,
                     bufB, We1t, b_e1, t, BATCH, H1_N, D_IN);
  hipLaunchKernelGGL(ln_relu_kernel, dim3(BATCH), blk, 0, stream, t, g_e1, be_e1, bufA, H1_N);
  hipLaunchKernelGGL(gemm_bias_kernel, dim3(H2_N / BN, BATCH / BM), blk, 0, stream,
                     bufA, We2t, b_e2, t, BATCH, H2_N, H1_N);
  hipLaunchKernelGGL(ln_relu_kernel, dim3(BATCH), blk, 0, stream, t, g_e2, be_e2, h2b, H2_N);

  // --- router ---
  hipLaunchKernelGGL(router_gemm_kernel, g1((size_t)BATCH * E_N), blk, 0, stream, h2b, Wrt, b_r, rlog);
  hipLaunchKernelGGL(router_epi_kernel, g1((size_t)BATCH), blk, 0, stream,
                     rlog, g_r, be_r, out + OFF_PROBS, t2v, t2i, BATCH);

  // --- mu / logvar heads (straight into d_out) ---
  hipLaunchKernelGGL(gemm_bias_kernel, dim3((E_N * L_N) / BN, BATCH / BM), blk, 0, stream,
                     h2b, Wmut, b_mu, out + OFF_MU, BATCH, E_N * L_N, H2_N);
  hipLaunchKernelGGL(gemm_bias_kernel, dim3((E_N * L_N) / BN, BATCH / BM), blk, 0, stream,
                     h2b, Wlvt, b_lv, out + OFF_LV, BATCH, E_N * L_N, H2_N);

  // --- reparameterize + combine ---
  hipLaunchKernelGGL(z_kernel, g1((size_t)BATCH * L_N), blk, 0, stream,
                     out + OFF_MU, out + OFF_LV, nois, t2v, t2i, zb);

  // --- decoder ---
  hipLaunchKernelGGL(gemm_bias_kernel, dim3(D1_N / BN, BATCH / BM), blk, 0, stream,
                     zb, Wd1t, b_d1, t, BATCH, D1_N, L_N);
  hipLaunchKernelGGL(ln_relu_kernel, dim3(BATCH), blk, 0, stream, t, g_d1, be_d1, bufB, D1_N);
  hipLaunchKernelGGL(gemm_bias_kernel, dim3(D2_N / BN, BATCH / BM), blk, 0, stream,
                     bufB, Wd2t, b_d2, t, BATCH, D2_N, D1_N);
  hipLaunchKernelGGL(ln_relu_kernel, dim3(BATCH), blk, 0, stream, t, g_d2, be_d2, bufA, D2_N);
  hipLaunchKernelGGL(gemm_bias_kernel, dim3(D_IN / BN, BATCH / BM), blk, 0, stream,
                     bufA, Wdot, b_do, out + OFF_RECON, BATCH, D_IN, D2_N);
}
